// Attention_79628693668242
// MI455X (gfx1250) — matrix operand order repaired and checked by reading, not compiled, not run on hardware
//
#include <hip/hip_runtime.h>

typedef unsigned short u16;
typedef unsigned int   u32;
typedef unsigned long long u64;
typedef __attribute__((ext_vector_type(16))) u16     v16u;
typedef __attribute__((ext_vector_type(16))) __bf16  v16bf;
typedef __attribute__((ext_vector_type(8)))  float   v8f;
typedef __attribute__((ext_vector_type(4)))  u32     u32x4;
typedef __attribute__((ext_vector_type(8)))  int     i32x8;
typedef __attribute__((ext_vector_type(4)))  int     i32x4;

#define N_EMBD   2048
#define N_HEAD   16
#define HEAD_DIM 128
#define BB       4
#define TT       2048
#define MTOT     (BB * TT)          // 8192
#define FQKV     (3 * N_EMBD)       // 6144
#define SA_H     12                 // self-attention heads

#if defined(__has_builtin)
#if __has_builtin(__builtin_amdgcn_tensor_load_to_lds)
#define HAVE_TDM 1
#endif
#endif
#if __has_include(<hip/amd_detail/amd_gfx1250_TDM.h>)
#define TDM_6ARG 1
#endif

static __device__ __forceinline__ u16 to_bf16(float f) {
    unsigned u = __builtin_bit_cast(unsigned, f);
    u += 0x7FFFu + ((u >> 16) & 1u);   // round to nearest even
    return (u16)(u >> 16);
}

union FragU { v16u v; uint4 q[2]; };

// two 16-byte chunks -> one 16-half fragment
static __device__ __forceinline__ v16u ld2(const u16* p0, const u16* p1) {
    FragU f;
    f.q[0] = *(const uint4*)p0;
    f.q[1] = *(const uint4*)p1;
    return f.v;
}

static __device__ __forceinline__ v8f wmma_bf16(v16u a, v16u b, v8f c) {
    return __builtin_amdgcn_wmma_f32_16x16x32_bf16(
        false, __builtin_bit_cast(v16bf, a),
        false, __builtin_bit_cast(v16bf, b),
        (short)0, c, false, false);
}

// async DMA: 16 bytes global -> LDS, tracked by ASYNCcnt (one instr per wave)
static __device__ __forceinline__ void async_copy_b128(void* lds, const void* g) {
    asm volatile("global_load_async_to_lds_b128 %0, %1, off"
                 :: "v"((unsigned)(size_t)lds),
                    "v"((u64)(size_t)g)
                 : "memory");
}
#define WAIT_ASYNC(n) asm volatile("s_wait_asynccnt " #n ::: "memory")

// ---------------------------------------------------------------------------
// fp32 -> bf16 elementwise convert
// ---------------------------------------------------------------------------
__global__ void k_cvt(const float* __restrict__ src, u16* __restrict__ dst, int n) {
    int i = blockIdx.x * blockDim.x + threadIdx.x;
    if (i < n) dst[i] = to_bf16(src[i]);
}

// copy-convert one head slice of prev_k/prev_v into the bf16 K/V buffers
__global__ void k_prev(const float* __restrict__ src, u16* __restrict__ dst, int h) {
    int i = blockIdx.x * blockDim.x + threadIdx.x;   // over B*T*D = 2^20
    if (i >= BB * TT * HEAD_DIM) return;
    int b = i >> 18;                 // T*D = 262144 = 2^18
    int r = i & 262143;
    size_t o = ((size_t)(b * N_HEAD + h) << 18) + r;
    dst[o] = to_bf16(src[o]);
}

// ---------------------------------------------------------------------------
// NT GEMM:  C[M,F] = A[M,K] * W[F,K]^T   (bf16 in, fp32 accumulate)
// block tile 128(M) x 64(F), 8 waves (4Mx2N), wave tile 32x32 (2x2 frags).
// LDS staging via async global->LDS DMA, double buffered (ASYNCcnt pipeline).
// mode 0: QKV epilogue (scatter q/k/v, write cur_k/cur_v fp32)
// mode 1: projection epilogue (fp32 y, F = 2048)
// ---------------------------------------------------------------------------
__global__ __launch_bounds__(256) void k_gemm(
    const u16* __restrict__ A, const u16* __restrict__ W, int K, int mode,
    u16* __restrict__ qb, u16* __restrict__ kbuf, u16* __restrict__ vbuf,
    float* __restrict__ ck, float* __restrict__ cv, float* __restrict__ yout) {

    __shared__ u16 As[2][128][32];
    __shared__ u16 Ws[2][64][32];

    const int tid  = threadIdx.x;
    const int lane = tid & 31;
    const int wave = tid >> 5;
    const int wm   = wave >> 1;      // 0..3
    const int wn   = wave & 1;       // 0..1
    const int g    = lane >> 4;      // half-wave group
    const int ln   = lane & 15;

    const int rowM = blockIdx.y * 128;
    const int colN = blockIdx.x * 64;

    v8f acc[2][2] = {};

    // stage one 32-deep k-slice into buffer `buf`
    // 768 x 16B transfers; each wave issues exactly 3 async instructions.
    auto stage = [&](int buf, int k0) {
        // j = 0,1 : A tile (512 transfers), j = 2 : W tile (256 transfers)
        for (int j = 0; j < 2; ++j) {
            int o = tid + 256 * j;
            int r = o >> 2, c = (o & 3) * 8;
            async_copy_b128(&As[buf][r][c], A + (size_t)(rowM + r) * K + k0 + c);
        }
        {
            int r = tid >> 2, c = (tid & 3) * 8;
            async_copy_b128(&Ws[buf][r][c], W + (size_t)(colN + r) * K + k0 + c);
        }
    };

    stage(0, 0);
    int buf = 0;
    for (int k0 = 0; k0 < K; k0 += 32, buf ^= 1) {
        const bool more = (k0 + 32) < K;
        if (more) stage(buf ^ 1, k0 + 32);     // prefetch next slice (3 ops/wave)
        if (more) { WAIT_ASYNC(0x3); }         // slice `buf` complete (in-order)
        else      { WAIT_ASYNC(0x0); }
        __syncthreads();

        // A fragments (16-bit 16x32 layout): lane m, chunks at k = g*8 and 16+g*8
        v16u a0 = ld2(&As[buf][wm * 32 +      ln][g * 8], &As[buf][wm * 32 +      ln][16 + g * 8]);
        v16u a1 = ld2(&As[buf][wm * 32 + 16 + ln][g * 8], &As[buf][wm * 32 + 16 + ln][16 + g * 8]);
        // B fragments (16-bit 32x16 layout): lane n, chunks at k = g*8 and 16+g*8
        v16u b0 = ld2(&Ws[buf][wn * 32 +      ln][g * 8], &Ws[buf][wn * 32 +      ln][16 + g * 8]);
        v16u b1 = ld2(&Ws[buf][wn * 32 + 16 + ln][g * 8], &Ws[buf][wn * 32 + 16 + ln][16 + g * 8]);

        acc[0][0] = wmma_bf16(a0, b0, acc[0][0]);
        acc[0][1] = wmma_bf16(a0, b1, acc[0][1]);
        acc[1][0] = wmma_bf16(a1, b0, acc[1][0]);
        acc[1][1] = wmma_bf16(a1, b1, acc[1][1]);
        __syncthreads();                       // readers done before buf is re-filled
    }

    // epilogue: C layout => row m = g*8 + i, col n = ln
    for (int fm = 0; fm < 2; ++fm)
    for (int fn = 0; fn < 2; ++fn)
    for (int i = 0; i < 8; ++i) {
        int grow = rowM + wm * 32 + fm * 16 + g * 8 + i;
        int gcol = colN + wn * 32 + fn * 16 + ln;
        float val = acc[fm][fn][i];
        if (mode == 1) {
            yout[(size_t)grow * N_EMBD + gcol] = val;
        } else {
            int b = grow >> 11, t = grow & 2047;
            if (gcol < N_EMBD) {                         // Q -> bf16 [B,H,T,D]
                int h = gcol >> 7, d = gcol & 127;
                qb[((size_t)(b * N_HEAD + h) * TT + t) * HEAD_DIM + d] = to_bf16(val);
            } else if (gcol < 2 * N_EMBD) {              // K
                int c = gcol - N_EMBD, h = c >> 7, d = c & 127;
                size_t idx = ((size_t)(b * N_HEAD + h) * TT + t) * HEAD_DIM + d;
                ck[idx]   = val;                         // cur_k fp32 output
                kbuf[idx] = to_bf16(val);
            } else {                                     // V
                int c = gcol - 2 * N_EMBD, h = c >> 7, d = c & 127;
                size_t idx = ((size_t)(b * N_HEAD + h) * TT + t) * HEAD_DIM + d;
                cv[idx]   = val;                         // cur_v fp32 output
                vbuf[idx] = to_bf16(val);
            }
        }
    }
}

// ---------------------------------------------------------------------------
// Flash attention, one (b, h, 64-query tile) per block; 4 waves x 16 rows.
// K tile staged by the Tensor Data Mover (TENSORcnt), V tile by async DMA
// (ASYNCcnt) + in-LDS transpose.  bf16 WMMA, fp32 online softmax.
// ---------------------------------------------------------------------------
__global__ __launch_bounds__(128) void k_attn(
    const u16* __restrict__ qb, const u16* __restrict__ kbuf,
    const u16* __restrict__ vbuf, u16* __restrict__ yb) {

    __shared__ u16 Ks[64][HEAD_DIM];     // K tile, row-major
    __shared__ u16 Vs[64][HEAD_DIM];     // V tile, row-major (DMA target)
    __shared__ u16 Vt[HEAD_DIM][64];     // V tile transposed: Vt[d][kd]
    __shared__ u16 Pws[4][16][64];       // per-wave P staging (C->A transpose)

    const int bh    = blockIdx.y;
    const int b     = bh >> 4;
    const int h     = bh & 15;
    const int qbase = blockIdx.x * 64;

    const int tid  = threadIdx.x;
    const int wave = tid >> 5;
    const int lane = tid & 31;
    const int g    = lane >> 4;
    const int ln   = lane & 15;
    const int qrow0 = qbase + wave * 16;

    const size_t headBase = (size_t)(b * N_HEAD + h) * TT * HEAD_DIM;

    // load Q fragments once (4 k-steps over D=128)
    const u16* qrow = qb + headBase + (size_t)(qrow0 + ln) * HEAD_DIM;
    v16u aQ[4];
    for (int ks = 0; ks < 4; ++ks)
        aQ[ks] = ld2(qrow + ks * 32 + g * 8, qrow + ks * 32 + 16 + g * 8);

    v8f   O[8] = {};
    float mrun[8], lrun[8];
    for (int i = 0; i < 8; ++i) { mrun[i] = -3.0e38f; lrun[i] = 0.0f; }

    const bool causal = (h < SA_H);
    const int  ntiles = causal ? (blockIdx.x + 1) : (TT / 64);
    const float scale = 0.08838834764831845f;   // 1/sqrt(128)

#ifdef HAVE_TDM
    // D# group-1: 2D tile, data_size=2B, 128-elem rows, stride 128, tile 128x64
    i32x8 g1;
    g1[0] = 0x00010000;          // workgroup_mask=0, data_size=1 (2 bytes)
    g1[1] = 128 << 16;           // tensor_dim0[15:0] = 128
    g1[2] = 64 << 16;            // tensor_dim0[31:16]=0 | tensor_dim1[15:0]=64
    g1[3] = 128 << 16;           // tensor_dim1[31:16]=0 | tile_dim0=128
    g1[4] = 64;                  // tile_dim1=64, tile_dim2=0
    g1[5] = 128;                 // tensor_dim0_stride = 128
    g1[6] = 0; g1[7] = 0;
    i32x4 gz = {0, 0, 0, 0};
    const u32 ksLds = (u32)(size_t)&Ks[0][0];
#endif

    for (int j = 0; j < ntiles; ++j) {
        const int kb0 = j * 64;

        // ---- stage K tile via Tensor Data Mover (wave 0 issues/waits) ----
#ifdef HAVE_TDM
        if (wave == 0) {
            u64 ga = (u64)(size_t)(kbuf + headBase + (size_t)kb0 * HEAD_DIM);
            u32x4 g0;
            g0[0] = 1u;                                  // count=1, user D#
            g0[1] = ksLds;                               // lds_addr
            g0[2] = (u32)ga;                             // global_addr[31:0]
            g0[3] = (u32)((ga >> 32) & 0x1FFFFFFu) | (2u << 30);  // addr hi | type=2
#ifdef TDM_6ARG
            __builtin_amdgcn_tensor_load_to_lds(g0, g1, gz, gz, g1, 0);
#else
            __builtin_amdgcn_tensor_load_to_lds(g0, g1, gz, gz, 0);
#endif
            __builtin_amdgcn_s_wait_tensorcnt(0);
        }
#else
        {   // fallback: async DMA for K tile (8 instr/wave)
            int kd = tid & 63, dh = (tid >> 6) * 64;
            const u16* kr = kbuf + headBase + (size_t)(kb0 + kd) * HEAD_DIM + dh;
            for (int q = 0; q < 8; ++q)
                async_copy_b128(&Ks[kd][dh + q * 8], kr + q * 8);
        }
#endif
        // ---- stage V tile via async DMA (8 instr/wave) ----
        {
            int kd = tid & 63, dh = (tid >> 6) * 64;
            const u16* vr = vbuf + headBase + (size_t)(kb0 + kd) * HEAD_DIM + dh;
            for (int q = 0; q < 8; ++q)
                async_copy_b128(&Vs[kd][dh + q * 8], vr + q * 8);
        }
        WAIT_ASYNC(0x0);
        __syncthreads();

        // ---- in-LDS transpose Vs -> Vt ----
        {
            int kd = tid & 63, dh = (tid >> 6) * 64;
            uint4 u[4];
            for (int q = 0; q < 4; ++q) u[q] = *(const uint4*)&Vs[kd][dh + q * 8];
            const u16* us = (const u16*)u;
            for (int c = 0; c < 64; ++c) Vt[dh + c][kd] = us[c];
        }
        __syncthreads();

        // ---- S = Q * K^T  (16 x 64 per wave, 4 n-chunks x 4 k-steps) ----
        v8f S[4] = {};
        for (int fn = 0; fn < 4; ++fn) {
            const u16* krow = &Ks[fn * 16 + ln][0];
            for (int ks = 0; ks < 4; ++ks) {
                v16u bK = ld2(krow + ks * 32 + g * 8, krow + ks * 32 + 16 + g * 8);
                S[fn] = wmma_bf16(aQ[ks], bK, S[fn]);
            }
        }

        // scale + causal mask
        const bool needmask = causal && (kb0 + 63 > qrow0);
        for (int fn = 0; fn < 4; ++fn)
            for (int i = 0; i < 8; ++i) {
                float s = S[fn][i] * scale;
                if (needmask && (kb0 + fn * 16 + ln > qrow0 + g * 8 + i)) s = -3.0e38f;
                S[fn][i] = s;
            }

        // online softmax: row max across 4 frags + 16 lanes of the group
        float pm[8], alpha[8], rs[8];
        for (int i = 0; i < 8; ++i)
            pm[i] = fmaxf(fmaxf(S[0][i], S[1][i]), fmaxf(S[2][i], S[3][i]));
        for (int m = 1; m <= 8; m <<= 1)
            for (int i = 0; i < 8; ++i)
                pm[i] = fmaxf(pm[i], __shfl_xor(pm[i], m, 32));
        for (int i = 0; i < 8; ++i) {
            float mn = fmaxf(mrun[i], pm[i]);
            alpha[i] = __expf(mrun[i] - mn);
            mrun[i]  = mn;
            rs[i]    = 0.0f;
        }
        for (int fn = 0; fn < 4; ++fn)
            for (int i = 0; i < 8; ++i) {
                float p = __expf(S[fn][i] - mrun[i]);
                S[fn][i] = p;
                rs[i] += p;
            }
        for (int m = 1; m <= 8; m <<= 1)
            for (int i = 0; i < 8; ++i)
                rs[i] += __shfl_xor(rs[i], m, 32);
        for (int i = 0; i < 8; ++i) lrun[i] = lrun[i] * alpha[i] + rs[i];
        for (int nc = 0; nc < 8; ++nc)
            for (int i = 0; i < 8; ++i) O[nc][i] *= alpha[i];

        // stage P (C layout -> row-major bf16) into this wave's LDS slab
        for (int fn = 0; fn < 4; ++fn)
            for (int i = 0; i < 8; ++i)
                Pws[wave][g * 8 + i][fn * 16 + ln] = to_bf16(S[fn][i]);
        __syncthreads();

        // ---- O += P * V  (2 k-steps over 64 keys, 8 n-chunks over D) ----
        for (int ks2 = 0; ks2 < 2; ++ks2) {
            v16u aP = ld2(&Pws[wave][ln][ks2 * 32 + g * 8],
                          &Pws[wave][ln][ks2 * 32 + 16 + g * 8]);
            for (int nc = 0; nc < 8; ++nc) {
                v16u bV = ld2(&Vt[nc * 16 + ln][ks2 * 32 + g * 8],
                              &Vt[nc * 16 + ln][ks2 * 32 + 16 + g * 8]);
                O[nc] = wmma_bf16(aP, bV, O[nc]);
            }
        }
        __syncthreads();   // tiles reused next iteration
    }

    // normalize + write y (bf16 [B,T,C]) for the projection GEMM
    float rl[8];
    for (int i = 0; i < 8; ++i) rl[i] = 1.0f / lrun[i];
    for (int nc = 0; nc < 8; ++nc)
        for (int i = 0; i < 8; ++i) {
            int t = qrow0 + g * 8 + i;
            int c = h * HEAD_DIM + nc * 16 + ln;
            yb[(size_t)(b * TT + t) * N_EMBD + c] = to_bf16(O[nc][i] * rl[i]);
        }
}

// ---------------------------------------------------------------------------
extern "C" void kernel_launch(void* const* d_in, const int* in_sizes, int n_in,
                              void* d_out, int out_size, void* d_ws, size_t ws_size,
                              hipStream_t stream) {
    const float* x      = (const float*)d_in[0];
    const float* w_attn = (const float*)d_in[1];
    const float* w_proj = (const float*)d_in[2];

    float* y_out = (float*)d_out;
    float* ck    = y_out + (size_t)MTOT * N_EMBD;                    // cur_k
    float* cv    = ck    + (size_t)BB * N_HEAD * TT * HEAD_DIM;      // cur_v

    // workspace carve-up (bf16 buffers)
    char* w = (char*)d_ws;
    u16* xb   = (u16*)w; w += (size_t)MTOT * N_EMBD * 2;             // 32 MB
    u16* wab  = (u16*)w; w += (size_t)FQKV * N_EMBD * 2;             // 24 MB
    u16* wpb  = (u16*)w; w += (size_t)N_EMBD * N_EMBD * 2;           //  8 MB
    u16* qb   = (u16*)w; w += (size_t)BB * N_HEAD * TT * HEAD_DIM * 2;
    u16* kbuf = (u16*)w; w += (size_t)BB * N_HEAD * TT * HEAD_DIM * 2;
    u16* vbuf = (u16*)w; w += (size_t)BB * N_HEAD * TT * HEAD_DIM * 2;
    u16* yb   = (u16*)w; w += (size_t)MTOT * N_EMBD * 2;

    // 1) bf16 converts
    {
        int n = MTOT * N_EMBD;
        k_cvt<<<(n + 255) / 256, 256, 0, stream>>>(x, xb, n);
        n = FQKV * N_EMBD;
        k_cvt<<<(n + 255) / 256, 256, 0, stream>>>(w_attn, wab, n);
        n = N_EMBD * N_EMBD;
        k_cvt<<<(n + 255) / 256, 256, 0, stream>>>(w_proj, wpb, n);
    }

    // 2) QKV GEMM (writes q/k/v bf16 buffers + cur_k/cur_v fp32)
    k_gemm<<<dim3(FQKV / 64, MTOT / 128), 256, 0, stream>>>(
        xb, wab, N_EMBD, 0, qb, kbuf, vbuf, ck, cv, nullptr);

    // 3) overwrite cross heads 12..15 of K/V with prev slices
    {
        int n = BB * TT * HEAD_DIM;
        for (int i = 0; i < 4; ++i) {
            const float* pk = (const float*)d_in[3 + 2 * i];
            const float* pv = (const float*)d_in[4 + 2 * i];
            k_prev<<<(n + 255) / 256, 256, 0, stream>>>(pk, kbuf, SA_H + i);
            k_prev<<<(n + 255) / 256, 256, 0, stream>>>(pv, vbuf, SA_H + i);
        }
    }

    // 4) attention (flash, WMMA, TDM + async staging)
    k_attn<<<dim3(TT / 64, BB * N_HEAD), 128, 0, stream>>>(qb, kbuf, vbuf, yb);

    // 5) output projection -> fp32 y
    k_gemm<<<dim3(N_EMBD / 64, MTOT / 128), 256, 0, stream>>>(
        yb, wpb, N_EMBD, 1, nullptr, nullptr, nullptr, nullptr, nullptr, y_out);
}
